// LSTMmodel_82111184765090
// MI455X (gfx1250) — matrix operand order repaired and checked by reading, not compiled, not run on hardware
//
#include <hip/hip_runtime.h>

// Problem constants (match reference): T=512, B=64, H=512, L=2
#define Tt   512
#define Bb   64
#define Hh   512
#define Ll   2
#define G4H  2048                    // 4*H
#define BH   (Bb * Hh)               // 32768
#define NNT  128                     // N tiles (2048 / 16)
#define FRAG_DW   256                // dwords per fragment (32 lanes * 8)
#define AMAT_DW   16384              // dwords per packed 64x512 A matrix (64 frags)
#define KT_B_DW   ((size_t)NNT * FRAG_DW)        // B dwords per K-tile = 32768
#define PASS_B_DW ((size_t)16 * NNT * FRAG_DW)   // B dwords per 512-K pass
#define LAYER_DW  ((size_t)32 * NNT * FRAG_DW)   // B dwords per layer (K=1024)

typedef __attribute__((ext_vector_type(16))) __bf16       v16bf;
typedef __attribute__((ext_vector_type(8)))  float        v8f;
typedef __attribute__((ext_vector_type(8)))  unsigned int v8u;

static __device__ __forceinline__ unsigned short f2bf(float f) {
  unsigned u = __builtin_bit_cast(unsigned, f);
  u += 0x7FFFu + ((u >> 16) & 1u);        // round-to-nearest-even
  return (unsigned short)(u >> 16);
}
static __device__ __forceinline__ unsigned pack2(float a, float b) {
  return (unsigned)f2bf(a) | ((unsigned)f2bf(b) << 16);
}
static __device__ __forceinline__ v16bf frag_cast(v8u u) {
  return __builtin_bit_cast(v16bf, u);
}
static __device__ __forceinline__ float sigm(float x) {
  return 1.0f / (1.0f + __expf(-x));
}

// ---------------------------------------------------------------------------
// Pack W_ih (K=0..511) | W_hh (K=512..1023), transposed, fp32->bf16, into
// per-lane WMMA B-fragment order: Bpack[l][kt][nt][lane][8 dw].
// B 32x16 bf16 layout: N = lane%16; lanes>=16 add 8 to K; dw v<4 packs K=2v,2v+1, v>=4 packs K=2v+8,2v+9.
// ---------------------------------------------------------------------------
__global__ __launch_bounds__(256) void lstm_pack_b(
    const float* __restrict__ W_ih,      // [L, 4H, H]
    const float* __restrict__ W_hh,      // [L, 4H, H]
    unsigned* __restrict__ Bpack) {
  const int wave = threadIdx.x >> 5;
  const int lane = threadIdx.x & 31;
  const int fragId = blockIdx.x * 8 + wave;      // 0 .. 8191
  const int l  = fragId >> 12;                   // / (32*128)
  const int kt = (fragId >> 7) & 31;
  const int nt = fragId & 127;

  const int n  = nt * 16 + (lane & 15);          // gate column 0..2047
  const int kb = kt * 32 + ((lane & 16) ? 8 : 0);

  const float* src;
  int kk;
  if (kb < Hh) { src = W_ih + ((size_t)(l * G4H + n)) * Hh; kk = kb; }
  else         { src = W_hh + ((size_t)(l * G4H + n)) * Hh; kk = kb - Hh; }

  float vals[16];
#pragma unroll
  for (int q = 0; q < 4; ++q) {
    float4 f = *(const float4*)&src[kk + q * 4 + (q >> 1) * 8];
    vals[q * 4 + 0] = f.x; vals[q * 4 + 1] = f.y;
    vals[q * 4 + 2] = f.z; vals[q * 4 + 3] = f.w;
  }
  v8u o;
#pragma unroll
  for (int v = 0; v < 8; ++v) o[v] = pack2(vals[2 * v], vals[2 * v + 1]);
  *(v8u*)(Bpack + ((((size_t)l * 32 + kt) * NNT + nt) * 32 + lane) * 8) = o;
}

// ---------------------------------------------------------------------------
// Pack nmat row-major [64 x 512] fp32 matrices into WMMA A-fragment layout:
// dst[mat][kt 0..15][m 0..3][lane][8 dw].
// A 16x32 bf16 layout: M = lane%16; lanes<16 hold K {0..7,16..23},
// lanes>=16 hold K {8..15,24..31}; dw v<4: K=base+2v; v>=4: K=base+16+2(v-4).
// Used for x ([T,B,H] = T matrices) and hx ([L,B,H] = L matrices).
// ---------------------------------------------------------------------------
__global__ __launch_bounds__(256) void lstm_pack_a(
    const float* __restrict__ src, unsigned* __restrict__ dst) {
  const int wave = threadIdx.x >> 5;
  const int lane = threadIdx.x & 31;
  const int f = blockIdx.x * 8 + wave;   // global fragment id
  const int mat = f >> 6;                // 64 frags per matrix
  const int r   = f & 63;
  const int ktl = r >> 2;
  const int m   = r & 3;
  const int hi  = (lane >> 4) & 1;
  const int row = m * 16 + (lane & 15);
  const int k0  = ktl * 32 + hi * 8;

  const float* s = src + (size_t)mat * BH + (size_t)row * Hh;
  float4 a0 = *(const float4*)&s[k0];
  float4 a1 = *(const float4*)&s[k0 + 4];
  float4 b0 = *(const float4*)&s[k0 + 16];
  float4 b1 = *(const float4*)&s[k0 + 20];
  v8u o;
  o[0] = pack2(a0.x, a0.y); o[1] = pack2(a0.z, a0.w);
  o[2] = pack2(a1.x, a1.y); o[3] = pack2(a1.z, a1.w);
  o[4] = pack2(b0.x, b0.y); o[5] = pack2(b0.z, b0.w);
  o[6] = pack2(b1.x, b1.y); o[7] = pack2(b1.z, b1.w);
  *(v8u*)(dst + (size_t)mat * AMAT_DW + ((size_t)r * 32 + lane) * 8) = o;
}

// ---------------------------------------------------------------------------
// 16 K-tiles of fragment loads + WMMA, everything pre-packed in global (L2).
// No LDS, no barriers: compiler is free to pipeline loads over the WMMAs.
// ---------------------------------------------------------------------------
static __device__ __forceinline__ void gemm_pass(
    const unsigned* __restrict__ aSrc,   // packed A (16 kt frags), this pass
    const unsigned* __restrict__ b0,     // B base (kt=pass start), lane folded
    const unsigned* __restrict__ b1,
    int m, int lane, v8f& acc0, v8f& acc1) {
#pragma unroll 4
  for (int ktl = 0; ktl < 16; ++ktl) {
    v16bf aF  = frag_cast(*(const v8u*)(aSrc + (((size_t)(ktl * 4 + m)) * 32 + lane) * 8));
    v16bf bF0 = frag_cast(*(const v8u*)(b0 + (size_t)ktl * KT_B_DW));
    v16bf bF1 = frag_cast(*(const v8u*)(b1 + (size_t)ktl * KT_B_DW));
    acc0 = __builtin_amdgcn_wmma_f32_16x16x32_bf16(false, aF, false, bF0,
                                                   (short)0, acc0, false, false);
    acc1 = __builtin_amdgcn_wmma_f32_16x16x32_bf16(false, aF, false, bF1,
                                                   (short)0, acc1, false, false);
  }
}

// ---------------------------------------------------------------------------
// One LSTM cell update for one layer at one timestep.
// gates[64 x 2048] = [inp | h_prev] (64x1024 bf16 frags) x Bpack (1024x2048).
// Block owns 16 hidden columns (+ their i/f/g/o gate columns); epilogue fuses
// bias + activations + c/h update and re-packs new h into A-fragment layout.
// Block: 256 threads (8 waves); wave = (M-block 0..3) x (gate-pair 0..1).
// ---------------------------------------------------------------------------
__global__ __launch_bounds__(256) void lstm_step(
    const unsigned* __restrict__ aPackX,  // packed input frags (16 kt)
    const unsigned* __restrict__ aPackH,  // packed prev hidden frags (16 kt)
    unsigned* __restrict__ hPackOut,      // packed new hidden frags (write)
    float* __restrict__ c,                // [B, H] cell state (in place)
    float* __restrict__ hfin,             // [B, H] fp32 copy of new hidden
    const unsigned* __restrict__ Bpack,   // this layer's packed weights
    const float* __restrict__ b_ih,       // [4H]
    const float* __restrict__ b_hh,       // [4H]
    float* __restrict__ out_seq) {        // d_out + t*B*H for layer 1, or null
  __shared__ float Gs[4 * 64 * 17];       // gate staging i,f,g,o
  __shared__ float Hs[64 * 16];           // new-h staging for bf16 pair packing

  const int tid  = threadIdx.x;
  const int lane = tid & 31;
  const int wave = tid >> 5;
  const int m    = wave & 3;            // M block (rows m*16..m*16+15)
  const int g0   = (wave >> 2) * 2;     // gate pair {g0, g0+1}
  const int hblk = blockIdx.x;          // hidden block 0..31
  const int hi   = (lane >> 4) & 1;
  const int col  = lane & 15;

  const int    nt0     = g0 * 32 + hblk;
  const size_t laneOff = (size_t)lane * 8;

  v8f acc0 = {}; v8f acc1 = {};

  // pass 0: K = 0..511 (layer input)
  gemm_pass(aPackX,
            Bpack + (size_t)nt0 * FRAG_DW + laneOff,
            Bpack + (size_t)(nt0 + 32) * FRAG_DW + laneOff,
            m, lane, acc0, acc1);
  // pass 1: K = 512..1023 (recurrent)
  gemm_pass(aPackH,
            Bpack + PASS_B_DW + (size_t)nt0 * FRAG_DW + laneOff,
            Bpack + PASS_B_DW + (size_t)(nt0 + 32) * FRAG_DW + laneOff,
            m, lane, acc0, acc1);

  // ---- spill accumulators: Gs[gate][row 0..63][hcol 0..15] ----
#pragma unroll
  for (int r = 0; r < 8; ++r) {
    const int row = m * 16 + (hi ? (8 + r) : r);
    Gs[((g0)     * 64 + row) * 17 + col] = acc0[r];
    Gs[((g0 + 1) * 64 + row) * 17 + col] = acc1[r];
  }
  __syncthreads();

  // ---- fused bias + activations + cell update ----
  const int hbase = hblk * 16;
#pragma unroll
  for (int it = 0; it < 4; ++it) {
    const int idx = tid + it * 256;     // (b, hc)
    const int b   = idx >> 4;
    const int hc  = idx & 15;
    const int n   = hbase + hc;
    float gi = Gs[(0 * 64 + b) * 17 + hc] + b_ih[n]          + b_hh[n];
    float gf = Gs[(1 * 64 + b) * 17 + hc] + b_ih[Hh + n]     + b_hh[Hh + n];
    float gg = Gs[(2 * 64 + b) * 17 + hc] + b_ih[2 * Hh + n] + b_hh[2 * Hh + n];
    float go = Gs[(3 * 64 + b) * 17 + hc] + b_ih[3 * Hh + n] + b_hh[3 * Hh + n];
    float cO = c[b * Hh + n];
    float cn = sigm(gf) * cO + sigm(gi) * tanhf(gg);
    float hn = sigm(go) * tanhf(cn);
    c[b * Hh + n]    = cn;
    hfin[b * Hh + n] = hn;
    if (out_seq) out_seq[b * Hh + n] = hn;
    Hs[b * 16 + hc] = hn;
  }
  __syncthreads();

  // ---- re-pack new h (bf16 pairs) into A-fragment layout for next step ----
#pragma unroll
  for (int it = 0; it < 2; ++it) {
    const int p  = tid + it * 256;      // pair index 0..511
    const int b  = p >> 3;              // row
    const int hc = (p & 7) * 2;         // even column in this block
    const unsigned dw = pack2(Hs[b * 16 + hc], Hs[b * 16 + hc + 1]);
    const int n   = hbase + hc;         // K index for next GEMM
    const int ktl = n >> 5;
    const int kp  = n & 31;
    const int h2  = (kp >> 3) & 1;
    const int j   = ((kp >> 4) & 1) * 4 + ((kp & 7) >> 1);
    const int ln  = h2 * 16 + (b & 15);
    const int mm  = b >> 4;
    hPackOut[(((size_t)ktl * 4 + mm) * 32 + ln) * 8 + j] = dw;
  }
}

// ---------------------------------------------------------------------------
extern "C" void kernel_launch(void* const* d_in, const int* in_sizes, int n_in,
                              void* d_out, int out_size, void* d_ws, size_t ws_size,
                              hipStream_t stream) {
  (void)in_sizes; (void)n_in; (void)out_size; (void)ws_size;
  const float* x    = (const float*)d_in[0];  // [T, B, H]
  const float* hx   = (const float*)d_in[1];  // [L, B, H]
  const float* cx   = (const float*)d_in[2];  // [L, B, H]
  const float* W_ih = (const float*)d_in[3];  // [L, 4H, H]
  const float* W_hh = (const float*)d_in[4];  // [L, 4H, H]
  const float* b_ih = (const float*)d_in[5];  // [L, 4H]
  const float* b_hh = (const float*)d_in[6];  // [L, 4H]
  float* out = (float*)d_out;                 // [T,B,H] ++ [L,B,H] ++ [L,B,H]

  // Workspace: c (f32) | h final (f32) | h packed ping-pong | Bpack | Xpack
  float*    cbuf  = (float*)d_ws;                        // L*BH floats
  float*    hfin  = cbuf + (size_t)Ll * BH;              // L*BH floats
  unsigned* hpack = (unsigned*)(hfin + (size_t)Ll * BH); // 2 * L * AMAT_DW
  unsigned* Bpack = hpack + 2 * (size_t)Ll * AMAT_DW;    // 2 * LAYER_DW
  unsigned* Xpack = Bpack + 2 * LAYER_DW;                // T * AMAT_DW

  hipMemcpyAsync(cbuf, cx, (size_t)Ll * BH * sizeof(float),
                 hipMemcpyDeviceToDevice, stream);

  lstm_pack_b<<<1024, 256, 0, stream>>>(W_ih, W_hh, Bpack);
  lstm_pack_a<<<Tt * 8, 256, 0, stream>>>(x, Xpack);       // all timesteps
  lstm_pack_a<<<Ll * 8, 256, 0, stream>>>(hx, hpack);      // initial h -> buf 0

  for (int t = 0; t < Tt; ++t) {
    unsigned* hp_r = hpack + (size_t)(t & 1) * Ll * AMAT_DW;
    unsigned* hp_w = hpack + (size_t)((t + 1) & 1) * Ll * AMAT_DW;
    // layer 0: input = packed x_t
    lstm_step<<<32, 256, 0, stream>>>(Xpack + (size_t)t * AMAT_DW,
                                      hp_r, hp_w, cbuf, hfin,
                                      Bpack, b_ih, b_hh, nullptr);
    // layer 1: input = layer-0 new hidden (just packed); writes output seq
    lstm_step<<<32, 256, 0, stream>>>(hp_w,
                                      hp_r + AMAT_DW, hp_w + AMAT_DW,
                                      cbuf + BH, hfin + BH,
                                      Bpack + LAYER_DW, b_ih + G4H, b_hh + G4H,
                                      out + (size_t)t * BH);
  }

  // tuple tail: hx_f then cx_f
  hipMemcpyAsync(out + (size_t)Tt * BH, hfin,
                 (size_t)Ll * BH * sizeof(float), hipMemcpyDeviceToDevice, stream);
  hipMemcpyAsync(out + (size_t)Tt * BH + (size_t)Ll * BH, cbuf,
                 (size_t)Ll * BH * sizeof(float), hipMemcpyDeviceToDevice, stream);
}
